// ASGCN_NEW_58935541236104
// MI455X (gfx1250) — matrix operand order repaired and checked by reading, not compiled, not run on hardware
//
#include <hip/hip_runtime.h>
#include <hip/hip_bf16.h>
#include <math.h>

// ---------------------------------------------------------------------------
// ASGCN forward for MI455X (gfx1250, wave32, WMMA).
// All dense math runs through v_wmma_f32_16x16x32_bf16 with LDS-staged tiles;
// A tiles are staged with GLOBAL_LOAD_ASYNC_TO_LDS_B128 (ASYNCcnt path).
// LSTM recurrence runs resident on one WGP per direction using 315KB of LDS.
// ---------------------------------------------------------------------------

typedef unsigned short ush;
typedef __bf16  bf16t;
typedef bf16t   v16bf  __attribute__((ext_vector_type(16)));
typedef float   v8f    __attribute__((ext_vector_type(8)));
typedef unsigned int u32x4 __attribute__((ext_vector_type(4)));

#define BB   32
#define SS   256
#define HH   512
#define EE   300
#define EP   320          // E padded to multiple of 32
#define BS   (BB*SS)      // 8192
#define KCAT 832          // 512 (h) + 300 (x) + 20 pad
#define G4H  2048

__device__ __forceinline__ ush f2bf(float f){
    unsigned u = __float_as_uint(f);
    unsigned r = u + 0x7FFFu + ((u >> 16) & 1u);
    return (ush)(r >> 16);
}
__device__ __forceinline__ float sigm(float x){ return 1.f/(1.f + __expf(-x)); }

union Frag { ush u[16]; u32x4 q[2]; v16bf v; };

// ---------------------------------------------------------------------------
// Generic tiled bf16 WMMA GEMM: C[M,N] = A[M,K] * B[K,N]  (row-major, batched)
// Block = 256 threads (8 waves), tile 64x128, K stepped by 32 through LDS.
// A tile staged via async global->LDS DMA (16B per lane, ASYNCcnt),
// B tile transposed through registers into per-column LDS layout so every
// WMMA fragment is 1-2 ds_load_b128s. Epilogue: optional per-row divide
// (GCN denom), bias[N], ReLU, f32 and/or bf16 outputs.
// M%64==0, N%128==0, K%32==0 by construction of all buffers.
// ---------------------------------------------------------------------------
__global__ __launch_bounds__(256) void gemm_bf16_kernel(
    const ush* __restrict__ A, long long sAb, int lda,
    const ush* __restrict__ B, long long sBb, int ldb,
    float* Cf, ush* Cbf, long long sCb, int ldc,
    int K,
    const float* __restrict__ bias,
    const float* __restrict__ rowdiv, int rdStride,
    int relu)
{
    __shared__ ush sA [64*32];     // A tile, row-major
    __shared__ ush sBt[128*32];    // B tile, transposed: sBt[col][k]

    const int bz = blockIdx.z;
    A += (size_t)bz * sAb;
    B += (size_t)bz * sBb;
    const size_t cOff = (size_t)bz * sCb;
    const int blockM = blockIdx.y * 64;
    const int blockN = blockIdx.x * 128;
    const int tid  = threadIdx.x;
    const int wave = tid >> 5, lane = tid & 31;
    const int wm = wave >> 2, wn = wave & 3;      // 2x4 wave grid, 32x32 each
    const int l15 = lane & 15;
    const int kh  = (lane >> 4) * 8;              // A-frag K-half select
    const int kb  = (lane >> 4) * 16;             // unused (B frags use kh)
    const int hi8 = (lane >> 4) * 8;              // C/D row select

    // Per-thread staging coordinates (fixed across K steps)
    const int aRow = tid >> 2, aC8 = (tid & 3) * 8;
    const unsigned ldsA = (unsigned)(size_t)&sA[aRow*32 + aC8];   // LDS byte offset
    const int bKr = tid >> 3, bCg = (tid & 7) * 16;

    v8f acc[2][2];
#pragma unroll
    for (int i = 0; i < 2; ++i)
#pragma unroll
        for (int j = 0; j < 2; ++j) acc[i][j] = (v8f){0,0,0,0,0,0,0,0};

    for (int kt = 0; kt < K; kt += 32) {
        {   // stage A: 64x32 bf16 via async DMA straight into LDS (16B/lane)
            const ush* g = A + (size_t)(blockM + aRow) * lda + kt + aC8;
            asm volatile("global_load_async_to_lds_b128 %0, %1, off"
                         :: "v"(ldsA), "v"((unsigned long long)(size_t)g)
                         : "memory");
            if (kt + 32 < K) __builtin_prefetch(g + 32, 0, 3);
        }
        {   // stage B transposed: 32x128 bf16 (two b128 loads, scatter by col)
            const ush* g = B + (size_t)(kt + bKr) * ldb + blockN + bCg;
            Frag t;
            t.q[0] = *(const u32x4*)g;
            t.q[1] = *(const u32x4*)(g + 8);
#pragma unroll
            for (int j = 0; j < 16; ++j) sBt[(bCg + j)*32 + bKr] = t.u[j];
            if (kt + 32 < K) __builtin_prefetch(g + (size_t)32*ldb, 0, 3);
        }
        // each wave drains its own async copies, then block-level barrier
        asm volatile("s_wait_asynccnt 0" ::: "memory");
        __syncthreads();

        Frag fa[2], fb[2];
#pragma unroll
        for (int mi = 0; mi < 2; ++mi) {
            int r = wm*32 + mi*16 + l15;
            fa[mi].q[0] = *(const u32x4*)&sA[r*32 + kh];
            fa[mi].q[1] = *(const u32x4*)&sA[r*32 + 16 + kh];
        }
#pragma unroll
        for (int ni = 0; ni < 2; ++ni) {
            int cn = wn*32 + ni*16 + l15;
            fb[ni].q[0] = *(const u32x4*)&sBt[cn*32 + kh];
            fb[ni].q[1] = *(const u32x4*)&sBt[cn*32 + 16 + kh];
        }
#pragma unroll
        for (int mi = 0; mi < 2; ++mi)
#pragma unroll
            for (int ni = 0; ni < 2; ++ni)
                acc[mi][ni] = __builtin_amdgcn_wmma_f32_16x16x32_bf16(
                    false, fa[mi].v, false, fb[ni].v,
                    (short)0, acc[mi][ni], false, false);
        __syncthreads();
    }

#pragma unroll
    for (int mi = 0; mi < 2; ++mi)
#pragma unroll
        for (int ni = 0; ni < 2; ++ni)
#pragma unroll
            for (int r = 0; r < 8; ++r) {
                int row = blockM + wm*32 + mi*16 + r + hi8;
                int col = blockN + wn*32 + ni*16 + l15;
                float v = acc[mi][ni][r];
                if (rowdiv) v /= rowdiv[bz*rdStride + row];
                if (bias)   v += bias[col];
                if (relu)   v = fmaxf(v, 0.f);
                size_t o = cOff + (size_t)row*ldc + col;
                if (Cf)  Cf[o]  = v;
                if (Cbf) Cbf[o] = f2bf(v);
            }
}

// ---------------------------------------------------------------------------
// BiLSTM recurrence: one block (32 waves) per direction, whole state in LDS.
// A_cat[32 x 832] bf16 = [h | x_t | pad]; gates[32 x 2048] f32.
// Per step: 32x2048x832 WMMA GEMM + gate nonlinearity (c in registers).
// ---------------------------------------------------------------------------
__global__ __launch_bounds__(1024) void lstm_kernel(
    const ush*  __restrict__ textbf,    // (BS, EP) bf16
    const ush*  __restrict__ wcatT,     // (2, 2048, 832) bf16, column-major B
    const float* __restrict__ bias_f,
    const float* __restrict__ bias_b,
    const int*  __restrict__ tidx,      // (B, S)
    float* __restrict__ Hout)           // (2, BS, HH) f32
{
    extern __shared__ char smem_raw[];
    ush*   Acat  = (ush*)smem_raw;                             // 32*832
    float* gates = (float*)(smem_raw + 32*KCAT*sizeof(ush));   // 32*2048

    const int dir  = blockIdx.x;
    const int tid  = threadIdx.x;
    const int wave = tid >> 5, lane = tid & 31;
    const int l15 = lane & 15;
    const int kh  = (lane >> 4) * 8;
    const int kb  = (lane >> 4) * 16;
    const int hi8 = (lane >> 4) * 8;
    const float* bias = dir ? bias_b : bias_f;
    const ush*   W    = wcatT + (size_t)dir * G4H * KCAT;

    float c[16];
#pragma unroll
    for (int j = 0; j < 16; ++j) c[j] = 0.f;
    for (int i = tid; i < 32*HH; i += 1024)
        Acat[(i >> 9)*KCAT + (i & 511)] = 0;     // h = 0

    for (int t = 0; t < SS; ++t) {
        const int ttv = dir ? (SS - 1 - t) : t;
        // phase 1: load x_t (incl. zero pad from padded textbf)
        for (int i = tid; i < 32*EP; i += 1024) {
            int b = i / EP, k = i - b*EP;
            Acat[b*KCAT + HH + k] = textbf[(size_t)(b*SS + ttv)*EP + k];
        }
        __syncthreads();
        // phase 2: gates = [h|x] @ Wcat  (256 16x16 tiles over 32 waves)
#pragma unroll 1
        for (int i = 0; i < 8; ++i) {
            int tile = wave*8 + i;
            int mt = tile & 1, nt = tile >> 1;
            v8f acc = (v8f){0,0,0,0,0,0,0,0};
            int rA = mt*16 + l15;
            int cn = nt*16 + l15;
            const ush* wcol = W + (size_t)cn * KCAT;
#pragma unroll 1
            for (int kk = 0; kk < KCAT; kk += 32) {
                Frag fa, fb;
                fa.q[0] = *(const u32x4*)&Acat[rA*KCAT + kk + kh];
                fa.q[1] = *(const u32x4*)&Acat[rA*KCAT + kk + 16 + kh];
                fb.q[0] = *(const u32x4*)&wcol[kk + kh];
                fb.q[1] = *(const u32x4*)&wcol[kk + 16 + kh];
                acc = __builtin_amdgcn_wmma_f32_16x16x32_bf16(
                    false, fa.v, false, fb.v, (short)0, acc, false, false);
            }
#pragma unroll
            for (int r = 0; r < 8; ++r)
                gates[(mt*16 + r + hi8)*G4H + nt*16 + l15] = acc[r];
        }
        __syncthreads();
        // phase 3: nonlinearity, h/c update, masked output
#pragma unroll 1
        for (int j = 0; j < 16; ++j) {
            int e = j*1024 + tid;
            int b = e >> 9, hh = e & 511;
            float iv = gates[b*G4H + hh]         + bias[hh];
            float fv = gates[b*G4H + 512  + hh]  + bias[512  + hh];
            float gv = gates[b*G4H + 1024 + hh]  + bias[1024 + hh];
            float ov = gates[b*G4H + 1536 + hh]  + bias[1536 + hh];
            float cn_ = sigm(fv)*c[j] + sigm(iv)*tanhf(gv);
            float hn  = sigm(ov)*tanhf(cn_);
            float outv = 0.f;
            if (tidx[b*SS + ttv] != 0) {
                c[j] = cn_;
                Acat[b*KCAT + hh] = f2bf(hn);
                outv = hn;
            }
            Hout[((size_t)dir*BS + b*SS + ttv)*HH + hh] = outv;
        }
        // no barrier needed here: next phase-1 writes a disjoint LDS region
        // and the barrier after phase 1 orders phase-3 writes vs phase-2 reads
    }
}

// ------------------------------ small kernels ------------------------------

__global__ void lens_kernel(const int* ti, const int* ai, const int* li, float* lens){
    int b = threadIdx.x;
    if (b < BB) {
        int tl = 0; for (int s = 0; s < SS; ++s) tl += (ti[b*SS + s] != 0);
        int al = 0; for (int s = 0; s < 4;  ++s) al += (ai[b*4  + s] != 0);
        int ll = 0; for (int s = 0; s < 16; ++s) ll += (li[b*16 + s] != 0);
        lens[b*8+0] = (float)tl;
        lens[b*8+1] = (float)al;
        lens[b*8+2] = (float)ll;
        lens[b*8+3] = (float)ll;              // a0
        lens[b*8+4] = (float)(ll + al - 1);   // a1
        lens[b*8+5] = (float)(tl - al);       // cl
    }
}

__global__ void embed_kernel(const int* ti, const float* emb, ush* out){
    int idx = blockIdx.x*256 + threadIdx.x;
    if (idx >= BS*EP) return;
    int bs = idx / EP, k = idx - bs*EP;
    float v = (k < EE) ? emb[(size_t)ti[bs]*EE + k] : 0.f;
    out[idx] = f2bf(v);
}

__global__ void wcat_pack_kernel(const float* wih_f, const float* whh_f,
                                 const float* wih_b, const float* whh_b, ush* out){
    long long idx = (long long)blockIdx.x*256 + threadIdx.x;
    if (idx >= 2LL*G4H*KCAT) return;
    int dir = (int)(idx / ((long long)G4H*KCAT));
    int rem = (int)(idx - (long long)dir*G4H*KCAT);
    int n = rem / KCAT, k = rem - n*KCAT;
    const float* wih = dir ? wih_b : wih_f;
    const float* whh = dir ? whh_b : whh_f;
    float v = 0.f;
    if (k < HH)            v = whh[(size_t)n*HH + k];
    else if (k < HH + EE)  v = wih[(size_t)n*EE + (k - HH)];
    out[idx] = f2bf(v);
}

__global__ void bfcvt_kernel(const float* in, ush* out, long long n){
    long long i = (long long)blockIdx.x*256 + threadIdx.x;
    if (i < n) out[i] = f2bf(in[i]);
}

__global__ void cwpack_kernel(const float* w, ush* out){  // (O=512,I=512,3) -> (1536,512)
    int idx = blockIdx.x*256 + threadIdx.x;
    if (idx >= 1536*512) return;
    int ko = idx >> 9, o = idx & 511;
    int t = ko / 512, i = ko - t*512;
    out[idx] = f2bf(w[((size_t)o*512 + i)*3 + t]);
}

__global__ void denom_kernel(const float* adj, float* denom){
    int idx = blockIdx.x*256 + threadIdx.x;
    if (idx >= BS) return;
    const float* row = adj + (size_t)idx * SS;
    float acc = 1.f;
    for (int t = 0; t < SS; ++t) acc += row[t];
    denom[idx] = acc;
}

__global__ void textout_kernel(const float* Hout, float* tof, ush* tobf, ush* tTbf){
    int idx = blockIdx.x*256 + threadIdx.x;
    if (idx >= BS*HH) return;
    int bs = idx >> 9, h = idx & 511;
    float v;
    if (h < 256) {
        const float* hf = Hout + (size_t)bs*HH;
        v = 0.5f*(hf[2*h] + hf[2*h + 1]);
    } else {
        const float* hb = Hout + (size_t)BS*HH + (size_t)bs*HH;
        int hh = h - 256;
        v = 0.5f*(hb[2*hh] + hb[2*hh + 1]);
    }
    tof[idx] = v;
    ush bv = f2bf(v);
    tobf[idx] = bv;
    int b = bs >> 8, s = bs & 255;
    tTbf[((size_t)b*HH + h)*SS + s] = bv;
}

// mode 0 = position weight, 1 = aspect mask, 2 = identity; optional +=
__global__ void rowweight_kernel(const float* x, float* yf, ush* ybf,
                                 const float* lens, int mode, int accum){
    long long idx = (long long)blockIdx.x*256 + threadIdx.x;
    if (idx >= (long long)BS*HH) return;
    int bs = (int)(idx >> 9);
    int b = bs >> 8;
    float s = (float)(bs & 255);
    float w = 1.f;
    if (mode == 0) {
        float a0 = lens[b*8+3], a1 = lens[b*8+4], tl = lens[b*8+0], cl = lens[b*8+5];
        if      (s <  a0) w = 1.f - (a0 - s)/cl;
        else if (s <= a1) w = 0.f;
        else if (s <  tl) w = 1.f - (s - a1)/cl;
        else              w = 0.f;
    } else if (mode == 1) {
        float a0 = lens[b*8+3], a1 = lens[b*8+4];
        w = (s >= a0 && s <= a1) ? 1.f : 0.f;
    }
    float v = w * x[idx];
    if (accum) { yf[idx] += v; }
    else {
        if (yf)  yf[idx]  = v;
        if (ybf) ybf[idx] = f2bf(v);
    }
}

__global__ void im2col_kernel(const float* x, ush* out){
    long long idx = (long long)blockIdx.x*256 + threadIdx.x;
    if (idx >= (long long)BS*1536) return;
    int bs = (int)(idx / 1536), ko = (int)(idx - (long long)bs*1536);
    int t = ko >> 9, i = ko & 511;
    int b = bs >> 8, s = bs & 255;
    int ss = s + t - 1;
    float v = (ss >= 0 && ss < SS) ? x[((size_t)b*SS + ss)*HH + i] : 0.f;
    out[idx] = f2bf(v);
}

__global__ void zero_kernel(float* p, long long n){
    long long i = (long long)blockIdx.x*256 + threadIdx.x;
    if (i < n) p[i] = 0.f;
}

// m = 0.01 * layernorm(sigmoid(a)) + m   (one block per row of 512)
__global__ __launch_bounds__(256) void mupdate_kernel(
    const float* a, float* mf, ush* mbf, const float* g, const float* bb)
{
    __shared__ float red[256];
    int row = blockIdx.x, tid = threadIdx.x;
    const float* ar = a + (size_t)row*HH;
    float v0 = sigm(ar[tid]);
    float v1 = sigm(ar[tid + 256]);
    red[tid] = v0 + v1; __syncthreads();
    for (int o = 128; o > 0; o >>= 1) { if (tid < o) red[tid] += red[tid + o]; __syncthreads(); }
    float mu = red[0] * (1.f/512.f); __syncthreads();
    float d0 = v0 - mu, d1 = v1 - mu;
    red[tid] = d0*d0 + d1*d1; __syncthreads();
    for (int o = 128; o > 0; o >>= 1) { if (tid < o) red[tid] += red[tid + o]; __syncthreads(); }
    float inv = rsqrtf(red[0] * (1.f/512.f) + 1e-12f);
    float l0 = d0*inv*g[tid]       + bb[tid];
    float l1 = d1*inv*g[tid + 256] + bb[tid + 256];
    size_t o0 = (size_t)row*HH + tid;
    float m0 = mf[o0]       + 0.01f*l0;
    float m1 = mf[o0 + 256] + 0.01f*l1;
    mf[o0] = m0; mf[o0 + 256] = m1;
    mbf[o0] = f2bf(m0); mbf[o0 + 256] = f2bf(m1);
}

__global__ void colsum_kernel(const float* am, float* out){
    int idx = blockIdx.x*256 + threadIdx.x;
    if (idx >= BS) return;
    int b = idx >> 8, t = idx & 255;
    const float* base = am + (size_t)b*SS*SS + t;
    float acc = 0.f;
    for (int s = 0; s < SS; ++s) acc += base[(size_t)s*SS];
    out[idx] = acc;
}

__global__ __launch_bounds__(256) void softmax_kernel(const float* in, float* out){
    __shared__ float red[256];
    int b = blockIdx.x, tid = threadIdx.x;
    float v = in[b*SS + tid];
    red[tid] = v; __syncthreads();
    for (int o = 128; o > 0; o >>= 1) { if (tid < o) red[tid] = fmaxf(red[tid], red[tid + o]); __syncthreads(); }
    float mx = red[0]; __syncthreads();
    float e = __expf(v - mx);
    red[tid] = e; __syncthreads();
    for (int o = 128; o > 0; o >>= 1) { if (tid < o) red[tid] += red[tid + o]; __syncthreads(); }
    out[b*SS + tid] = e / red[0];
}

__global__ void outvec_kernel(const float* w, const float* tof, float* fnout, int coloff){
    int idx = blockIdx.x*256 + threadIdx.x;
    if (idx >= BB*HH) return;
    int b = idx >> 9, h = idx & 511;
    float acc = 0.f;
    for (int t = 0; t < SS; ++t)
        acc += w[b*SS + t] * tof[((size_t)b*SS + t)*HH + h];
    fnout[b*1536 + coloff + h] = acc;
}

__global__ void fc_kernel(const float* fn, const float* w, const float* bias, float* out){
    int tid = threadIdx.x;
    if (tid >= BB*3) return;
    int b = tid / 3, p = tid - b*3;
    float acc = bias[p];
    for (int k = 0; k < 1536; ++k) acc += fn[b*1536 + k] * w[p*1536 + k];
    out[tid] = acc;
}

// ------------------------------- launcher ----------------------------------

extern "C" void kernel_launch(void* const* d_in, const int* in_sizes, int n_in,
                              void* d_out, int out_size, void* d_ws, size_t ws_size,
                              hipStream_t stream)
{
    (void)in_sizes; (void)n_in; (void)out_size; (void)ws_size;
    const int*   ti    = (const int*)  d_in[0];
    const int*   ai    = (const int*)  d_in[1];
    const int*   li    = (const int*)  d_in[2];
    const float* adj   = (const float*)d_in[3];
    const float* emb   = (const float*)d_in[4];
    const float* wih_f = (const float*)d_in[5];
    const float* whh_f = (const float*)d_in[6];
    const float* b_f   = (const float*)d_in[7];
    const float* wih_b = (const float*)d_in[8];
    const float* whh_b = (const float*)d_in[9];
    const float* b_b   = (const float*)d_in[10];
    const float* gcw   = (const float*)d_in[11];
    const float* gcb   = (const float*)d_in[12];
    const float* c1w   = (const float*)d_in[13];
    const float* c1b   = (const float*)d_in[14];
    const float* c2w   = (const float*)d_in[15];
    const float* c2b   = (const float*)d_in[16];
    const float* fcw   = (const float*)d_in[17];
    const float* fcb   = (const float*)d_in[18];
    const float* lng   = (const float*)d_in[19];
    const float* lnb   = (const float*)d_in[20];
    float* out = (float*)d_out;

    char* wsb = (char*)d_ws;
    size_t off = 0;
    auto alloc = [&](size_t bytes)->char* {
        char* p = wsb + off;
        off = (off + bytes + 255) & ~(size_t)255;
        return p;
    };
    float* lens   = (float*)alloc((size_t)BB*8*4);
    ush*   textbf = (ush*)  alloc((size_t)BS*EP*2);
    ush*   wcatT  = (ush*)  alloc((size_t)2*G4H*KCAT*2);
    float* Hout   = (float*)alloc((size_t)2*BS*HH*4);   // later: xc1 | xc2 | afbuf
    float* tof    = (float*)alloc((size_t)BS*HH*4);
    ush*   tobf   = (ush*)  alloc((size_t)BS*HH*2);
    ush*   tTbf   = (ush*)  alloc((size_t)BB*HH*SS*2);
    ush*   adjbf  = (ush*)  alloc((size_t)BB*SS*SS*2);
    float* denom  = (float*)alloc((size_t)BS*4);
    ush*   gcwbf  = (ush*)  alloc((size_t)6*HH*HH*2);
    ush*   cw1bf  = (ush*)  alloc((size_t)1536*HH*2);
    ush*   cw2bf  = (ush*)  alloc((size_t)1536*HH*2);
    float* pwf    = (float*)alloc((size_t)BS*HH*4);
    ush*   pwbf   = (ush*)  alloc((size_t)BS*HH*2);
    ush*   ic     = (ush*)  alloc((size_t)BS*1536*2);   // later: alpha_bf | alphaM
    ush*   hbufbf = (ush*)  alloc((size_t)BS*HH*2);
    float* mf     = (float*)alloc((size_t)BS*HH*4);     // graph_mask / m
    ush*   mbf    = (ush*)  alloc((size_t)BS*HH*2);
    float* csum   = (float*)alloc((size_t)BS*4);
    float* smx    = (float*)alloc((size_t)BS*4);
    float* fnout  = (float*)alloc((size_t)BB*1536*4);

    ush*   alphabf = ic;                                    // 32*256*256 bf16
    float* alphaM  = (float*)(ic + (size_t)BB*SS*SS);       // 32*256*256 f32
    float* xc1   = Hout;
    float* xc2   = Hout + (size_t)BS*HH;
    float* afbuf = Hout;                                    // free after conv1

    auto gemm = [&](const ush* A, long long sAb, int lda,
                    const ush* Bm, long long sBb, int ldb,
                    float* Cf, ush* Cbf, long long sCb, int ldc,
                    int M, int N, int K, int batches,
                    const float* bias, const float* rowdiv, int rdStride, int relu){
        dim3 g(N/128, M/64, batches);
        gemm_bf16_kernel<<<g, 256, 0, stream>>>(A, sAb, lda, Bm, sBb, ldb,
                                                Cf, Cbf, sCb, ldc, K,
                                                bias, rowdiv, rdStride, relu);
    };
    auto rw = [&](const float* x, float* yf, ush* ybf, int mode, int accum){
        rowweight_kernel<<<(int)(((long long)BS*HH + 255)/256), 256, 0, stream>>>(
            x, yf, ybf, lens, mode, accum);
    };

    // ---- preprocessing / packing ----
    lens_kernel<<<1, 64, 0, stream>>>(ti, ai, li, lens);
    embed_kernel<<<(BS*EP + 255)/256, 256, 0, stream>>>(ti, emb, textbf);
    wcat_pack_kernel<<<(int)((2LL*G4H*KCAT + 255)/256), 256, 0, stream>>>(
        wih_f, whh_f, wih_b, whh_b, wcatT);
    bfcvt_kernel<<<(6*HH*HH + 255)/256, 256, 0, stream>>>(gcw, gcwbf, 6LL*HH*HH);
    cwpack_kernel<<<(1536*512 + 255)/256, 256, 0, stream>>>(c1w, cw1bf);
    cwpack_kernel<<<(1536*512 + 255)/256, 256, 0, stream>>>(c2w, cw2bf);
    bfcvt_kernel<<<(int)(((long long)BB*SS*SS + 255)/256), 256, 0, stream>>>(
        adj, adjbf, (long long)BB*SS*SS);
    denom_kernel<<<(BS + 255)/256, 256, 0, stream>>>(adj, denom);

    // ---- BiLSTM (resident, 315KB LDS per direction) ----
    size_t lstm_lds = (size_t)32*KCAT*sizeof(ush) + (size_t)32*G4H*sizeof(float);
    lstm_kernel<<<2, 1024, lstm_lds, stream>>>(textbf, wcatT, b_f, b_b, ti, Hout);
    textout_kernel<<<(BS*HH + 255)/256, 256, 0, stream>>>(Hout, tof, tobf, tTbf);

    // ---- conv path: x_conv = relu(conv2(pw(relu(conv1(pw(text_out)))))) ----
    rw(tof, pwf, pwbf, 0, 0);                                 // pw(text_out) f32+bf16
    im2col_kernel<<<(int)(((long long)BS*1536 + 255)/256), 256, 0, stream>>>(pwf, ic);
    gemm(ic, 0, 1536, cw1bf, 0, 512, xc1, nullptr, 0, 512,
         BS, 512, 1536, 1, c1b, nullptr, 0, 1);
    rw(xc1, pwf, nullptr, 0, 0);                              // pw(x_conv1), f32 only
    im2col_kernel<<<(int)(((long long)BS*1536 + 255)/256), 256, 0, stream>>>(pwf, ic);
    gemm(ic, 0, 1536, cw2bf, 0, 512, xc2, nullptr, 0, 512,
         BS, 512, 1536, 1, c2b, nullptr, 0, 1);

    // ---- GCN stack ----
    // x = relu(gcn0(pw(text_out)))
    gemm(pwbf, 0, 512, gcwbf, 0, 512, nullptr, hbufbf, 0, 512,
         BS, 512, 512, 1, nullptr, nullptr, 0, 0);
    gemm(adjbf, (long long)SS*SS, SS, hbufbf, (long long)SS*HH, HH,
         pwf, nullptr, (long long)SS*HH, HH, SS, HH, SS, BB,
         gcb, denom, SS, 1);
    rw(pwf, nullptr, pwbf, 0, 0);                             // pwx -> bf16 only
    zero_kernel<<<(int)(((long long)BS*HH + 255)/256), 256, 0, stream>>>(
        mf, (long long)BS*HH);
    for (int k = 1; k < 6; ++k) {
        gemm(pwbf, 0, 512, gcwbf + (size_t)k*HH*HH, 0, 512,
             nullptr, hbufbf, 0, 512, BS, 512, 512, 1, nullptr, nullptr, 0, 0);
        gemm(adjbf, (long long)SS*SS, SS, hbufbf, (long long)SS*HH, HH,
             pwf, nullptr, (long long)SS*HH, HH, SS, HH, SS, BB,
             gcb + (size_t)k*HH, denom, SS, 1);
        rw(pwf, mf, nullptr, 1, 1);                           // graph_mask += mask(.)
    }

    // ---- three 10-hop attends ----
    auto attend = [&](int lnidx, int coloff){
        for (int it = 0; it < 9; ++it) {
            gemm(mbf, (long long)SS*HH, HH, tTbf, (long long)HH*SS, SS,
                 nullptr, alphabf, (long long)SS*SS, SS,
                 SS, SS, HH, BB, nullptr, nullptr, 0, 0);
            gemm(alphabf, (long long)SS*SS, SS, tobf, (long long)SS*HH, HH,
                 afbuf, nullptr, (long long)SS*HH, HH,
                 SS, HH, SS, BB, nullptr, nullptr, 0, 0);
            mupdate_kernel<<<BS, 256, 0, stream>>>(afbuf, mf, mbf,
                                                   lng + (size_t)lnidx*HH,
                                                   lnb + (size_t)lnidx*HH);
        }
        gemm(mbf, (long long)SS*HH, HH, tTbf, (long long)HH*SS, SS,
             alphaM, nullptr, (long long)SS*SS, SS,
             SS, SS, HH, BB, nullptr, nullptr, 0, 0);
        colsum_kernel<<<(BS + 255)/256, 256, 0, stream>>>(alphaM, csum);
        softmax_kernel<<<BB, 256, 0, stream>>>(csum, smx);
        outvec_kernel<<<(BB*HH + 255)/256, 256, 0, stream>>>(smx, tof, fnout, coloff);
    };

    rw(mf, nullptr, mbf, 2, 0);          // m1 = graph_mask (mf already holds it)
    attend(0, 0);
    rw(tof, mf, mbf, 1, 0);              // m2 = mask(text_out)
    attend(1, 512);
    rw(xc2, mf, mbf, 1, 0);              // m3 = mask(x_conv)
    attend(2, 1024);

    fc_kernel<<<1, 128, 0, stream>>>(fnout, fcw, fcb, out);
}
